// ScaledDotProductAttention_32633161515353
// MI455X (gfx1250) — matrix operand order repaired and checked by reading, not compiled, not run on hardware
//
#include <hip/hip_runtime.h>
#include <math.h>

typedef __attribute__((ext_vector_type(16))) __bf16 v16bf;
typedef __attribute__((ext_vector_type(8)))  __bf16 v8bf;
typedef __attribute__((ext_vector_type(8)))  float  v8f;
typedef __attribute__((ext_vector_type(4)))  float  v4f;

#define SEQ   2048
#define DH    64
#define BH    64         // B*H
#define NW    4          // waves per block
#define BM    (16 * NW)  // 64 q rows per block
#define BN    64         // kv tile (keys) for main kernel
#define PSTR  72         // sP row stride (bf16), 144B: 16B-aligned rows

// ---------------- 16-lane xor reduction: v_permlane16 if available ----------
#if defined(__has_builtin)
#  if __has_builtin(__builtin_amdgcn_permlane16)
#    define HAVE_PERMLANE16 1
#  endif
#endif

constexpr unsigned long long xsel(int m) {
    unsigned long long s = 0;
    for (int i = 0; i < 16; ++i)
        s |= (unsigned long long)((unsigned)((i ^ m) & 15)) << (4 * i);
    return s;
}

template<int M>
__device__ __forceinline__ float lanexor(float v) {
#if defined(HAVE_PERMLANE16)
    constexpr unsigned long long s = xsel(M);
    unsigned u = __builtin_bit_cast(unsigned, v);
    u = __builtin_amdgcn_permlane16(u, u, (unsigned)(s & 0xffffffffu),
                                    (unsigned)(s >> 32), false, false);
    return __builtin_bit_cast(float, u);
#else
    return __shfl_xor(v, M, 32);
#endif
}

__device__ __forceinline__ float redmax16(float v) {
    v = fmaxf(v, lanexor<1>(v));
    v = fmaxf(v, lanexor<2>(v));
    v = fmaxf(v, lanexor<4>(v));
    v = fmaxf(v, lanexor<8>(v));
    return v;
}
__device__ __forceinline__ float redsum16(float v) {
    v += lanexor<1>(v);
    v += lanexor<2>(v);
    v += lanexor<4>(v);
    v += lanexor<8>(v);
    return v;
}

// ---------------- prepass: K fp32 -> bf16 (row major) -----------------------
__global__ __launch_bounds__(256)
void conv_k(const float* __restrict__ in, __bf16* __restrict__ out) {
    const size_t i = ((size_t)blockIdx.x * blockDim.x + threadIdx.x) * 8;
    v4f a = *(const v4f*)(in + i);
    v4f b = *(const v4f*)(in + i + 4);
    v8bf r;
    #pragma unroll
    for (int j = 0; j < 4; ++j) { r[j] = (__bf16)a[j]; r[4 + j] = (__bf16)b[j]; }
    *(v8bf*)(out + i) = r;
}

// ---------------- prepass: V fp32 [s][d] -> bf16 VT [d][s] ------------------
__global__ __launch_bounds__(256)
void conv_vt(const float* __restrict__ V, __bf16* __restrict__ VT) {
    __shared__ __bf16 tile[64 * 72];
    const int bh = blockIdx.y;
    const int s0 = blockIdx.x * 64;
    const float* Vb = V + (size_t)bh * SEQ * DH;
    __bf16* Tb = VT + (size_t)bh * DH * SEQ;
    const int t = threadIdx.x;
    {   // coalesced read, tiled store
        const int r = t >> 2, c4 = (t & 3) << 4;
        const float* vr = Vb + (size_t)(s0 + r) * DH + c4;
        #pragma unroll
        for (int i = 0; i < 16; ++i) tile[r * 72 + c4 + i] = (__bf16)vr[i];
    }
    __syncthreads();
    {   // transposed read, coalesced write
        const int d = t >> 2, sg = (t & 3) << 4;
        v8bf w0, w1;
        #pragma unroll
        for (int i = 0; i < 8; ++i) {
            w0[i] = tile[(sg + i)     * 72 + d];
            w1[i] = tile[(sg + 8 + i) * 72 + d];
        }
        __bf16* op = Tb + (size_t)d * SEQ + s0 + sg;
        *(v8bf*)(op)     = w0;
        *(v8bf*)(op + 8) = w1;
    }
}

// ---------------- main flash-attention kernel (bf16 K, VT in global) --------
struct WaveCtx {
    const __bf16* Kb;
    const __bf16* VTb;
    int qb;
    int am, ah, bn, bk, cm0;
};

template<bool MASKED>
__device__ __forceinline__ void attn_tile(const WaveCtx& w, int kb,
                                          const v16bf qa[2], v8f o[4],
                                          float mrow[8], float lrow[8],
                                          __bf16* pw) {
    const float cf = 0.125f * 1.44269504088896340736f; // 1/sqrt(64)*log2(e)

    // ---- GEMM1: S(16x64) = Q(16x64) x K^T ----
    v8f sc[4];
    #pragma unroll
    for (int c = 0; c < 4; ++c) {
        v8f acc;
        #pragma unroll
        for (int r = 0; r < 8; ++r) acc[r] = 0.0f;
        #pragma unroll
        for (int dstep = 0; dstep < 2; ++dstep) {
            const __bf16* kp = w.Kb + (size_t)(kb + 16 * c + w.bn) * DH
                                    + dstep * 32 + w.bk;
            v8bf b0 = *(const v8bf*)kp;
            v8bf b1 = *(const v8bf*)(kp + 16);
            v16bf bf = __builtin_shufflevector(b0, b1,
                0,1,2,3,4,5,6,7,8,9,10,11,12,13,14,15);
            acc = __builtin_amdgcn_wmma_f32_16x16x32_bf16(
                false, qa[dstep], false, bf, (short)0, acc, false, false);
        }
        if (MASKED) {
            const int key = kb + 16 * c + w.bn;
            #pragma unroll
            for (int r = 0; r < 8; ++r)
                if (key > w.qb + w.cm0 + r) acc[r] = -INFINITY;
        }
        sc[c] = acc;
    }

    // ---- row max (permlane reduction) ----
    float mnew[8], alpha[8];
    #pragma unroll
    for (int r = 0; r < 8; ++r) {
        float v = fmaxf(fmaxf(sc[0][r], sc[1][r]), fmaxf(sc[2][r], sc[3][r]));
        v = redmax16(v);
        mnew[r]  = fmaxf(mrow[r], v);
        alpha[r] = exp2f((mrow[r] - mnew[r]) * cf);
    }

    // ---- P = exp2((S - m)*cf), C-layout -> LDS as bf16 ----
    #pragma unroll
    for (int r = 0; r < 8; ++r) {
        #pragma unroll
        for (int c = 0; c < 4; ++c) {
            float p = exp2f((sc[c][r] - mnew[r]) * cf);
            pw[(w.cm0 + r) * PSTR + 16 * c + w.bn] = (__bf16)p;
        }
    }
    #pragma unroll
    for (int r = 0; r < 8; ++r) mrow[r] = mnew[r];
    #pragma unroll
    for (int j = 0; j < 4; ++j)
        #pragma unroll
        for (int r = 0; r < 8; ++r) o[j][r] *= alpha[r];

    // wave-private LDS RAW fence (cross-lane, same wave)
    asm volatile("s_wait_dscnt 0" ::: "memory");

    // ---- reload P as two 16x32 bf16 A-fragments ----
    v16bf pa[2];
    #pragma unroll
    for (int a = 0; a < 2; ++a) {
        const __bf16* pr = pw + w.am * PSTR + 32 * a;
        v8bf lo = *(const v8bf*)(pr + w.ah);
        v8bf hi = *(const v8bf*)(pr + 16 + w.ah);
        pa[a] = __builtin_shufflevector(lo, hi,
            0,1,2,3,4,5,6,7,8,9,10,11,12,13,14,15);
    }

    // ---- row sum of P via WMMA with ones-matrix (matrix pipe, not VALU) ----
    // D[m][n] = sum_k P[m][k] * 1 : per-row sum replicated across lanes,
    // in exactly the C-fragment row mapping lrow uses.
    {
        v16bf ones;
        #pragma unroll
        for (int e = 0; e < 16; ++e) ones[e] = (__bf16)1.0f;
        v8f ps;
        #pragma unroll
        for (int r = 0; r < 8; ++r) ps[r] = 0.0f;
        ps = __builtin_amdgcn_wmma_f32_16x16x32_bf16(
            false, pa[0], false, ones, (short)0, ps, false, false);
        ps = __builtin_amdgcn_wmma_f32_16x16x32_bf16(
            false, pa[1], false, ones, (short)0, ps, false, false);
        #pragma unroll
        for (int r = 0; r < 8; ++r)
            lrow[r] = lrow[r] * alpha[r] + ps[r];
    }

    // ---- GEMM2: O(16x64) += P(16x64) x V(64x64), via VT rows ----
    #pragma unroll
    for (int j = 0; j < 4; ++j) {
        #pragma unroll
        for (int a = 0; a < 2; ++a) {
            const __bf16* vp = w.VTb + (size_t)(16 * j + w.bn) * SEQ
                                     + kb + 32 * a + w.bk;
            v8bf b0 = *(const v8bf*)vp;
            v8bf b1 = *(const v8bf*)(vp + 16);
            v16bf vb = __builtin_shufflevector(b0, b1,
                0,1,2,3,4,5,6,7,8,9,10,11,12,13,14,15);
            o[j] = __builtin_amdgcn_wmma_f32_16x16x32_bf16(
                false, pa[a], false, vb, (short)0, o[j], false, false);
        }
    }
}

__global__ __launch_bounds__(32 * NW)
void fa_fwd_l2(const float* __restrict__ Qg, const __bf16* __restrict__ Kg,
               const __bf16* __restrict__ VTg, float* __restrict__ Og)
{
    __shared__ __bf16 sP[NW * 16 * PSTR]; // per-wave P bounce

    const int tid  = threadIdx.x;
    const int wid  = tid >> 5;
    const int lane = tid & 31;

    const int bh = blockIdx.y;
    const int bx = blockIdx.x;
    const int qb = bx * BM + wid * 16;

    WaveCtx w;
    w.Kb  = Kg  + (size_t)bh * SEQ * DH;
    w.VTb = VTg + (size_t)bh * DH * SEQ;
    w.qb  = qb;
    w.am  = lane & 15;
    w.ah  = (lane >> 4) << 3;
    w.bn  = lane & 15;
    w.bk  = (lane >> 4) << 3;
    w.cm0 = (lane >> 4) << 3;

    const float* Qb = Qg + (size_t)bh * SEQ * DH;
    float*       Ob = Og + (size_t)bh * SEQ * DH;
    __bf16* pw = sP + wid * 16 * PSTR;

    // ---- Q tile once: two 16x32 bf16 A-fragments ----
    v16bf qa[2];
    #pragma unroll
    for (int dstep = 0; dstep < 2; ++dstep) {
        const float* qr = Qb + (size_t)(qb + w.am) * DH + dstep * 32;
        #pragma unroll
        for (int e = 0; e < 16; ++e) {
            const int k = w.ah + (e & 7) + ((e >> 3) << 4);
            qa[dstep][e] = (__bf16)qr[k];
        }
    }

    v8f   o[4];
    float mrow[8], lrow[8];
    #pragma unroll
    for (int j = 0; j < 4; ++j)
        #pragma unroll
        for (int r = 0; r < 8; ++r) o[j][r] = 0.0f;
    #pragma unroll
    for (int r = 0; r < 8; ++r) { mrow[r] = -INFINITY; lrow[r] = 0.0f; }

    // tiles 0..bx-1 are provably unmasked; tile bx is the causal boundary
    for (int t = 0; t < bx; ++t) {
        const int kb = t * BN;
        __builtin_prefetch(w.Kb  + (size_t)(kb + BN + lane) * DH, 0, 1);
        __builtin_prefetch(w.VTb + (size_t)lane * SEQ + kb + BN, 0, 1);
        attn_tile<false>(w, kb, qa, o, mrow, lrow, pw);
    }
    attn_tile<true>(w, bx * BN, qa, o, mrow, lrow, pw);

    // ---- epilogue: O * (1/l) ----
    float invl[8];
    #pragma unroll
    for (int r = 0; r < 8; ++r) invl[r] = 1.0f / lrow[r];
    #pragma unroll
    for (int j = 0; j < 4; ++j)
        #pragma unroll
        for (int r = 0; r < 8; ++r)
            Ob[(size_t)(qb + w.cm0 + r) * DH + 16 * j + w.bn] = o[j][r] * invl[r];
}

// ---------------- fallback kernel (no workspace): round-1 design ------------
#define FKSTR 72
#define FVSTR 40
#define FPSTR 36

__global__ __launch_bounds__(32 * NW)
void fa_fwd_fallback(const float* __restrict__ Qg, const float* __restrict__ Kg,
                     const float* __restrict__ Vg, float* __restrict__ Og)
{
    __shared__ __bf16 sK [32 * FKSTR];
    __shared__ __bf16 sVT[DH * FVSTR];
    __shared__ float  sP [NW * 16 * FPSTR];

    const int tid  = threadIdx.x;
    const int wid  = tid >> 5;
    const int lane = tid & 31;

    const int bh     = blockIdx.y;
    const int blk_qb = blockIdx.x * BM;
    const int qb     = blk_qb + wid * 16;

    const size_t base = (size_t)bh * SEQ * DH;
    const float* Qb = Qg + base;
    const float* Kb = Kg + base;
    const float* Vb = Vg + base;
    float*       Ob = Og + base;

    const int am  = lane & 15;
    const int ah  = (lane >> 4) << 3;
    const int bn  = lane & 15;
    const int bk  = (lane >> 4) << 3;
    const int cm0 = (lane >> 4) << 3;

    const float cf = 0.125f * 1.44269504088896340736f;

    v16bf qa[2];
    #pragma unroll
    for (int dstep = 0; dstep < 2; ++dstep) {
        const float* qr = Qb + (size_t)(qb + am) * DH + dstep * 32;
        #pragma unroll
        for (int e = 0; e < 16; ++e) {
            const int k = ah + (e & 7) + ((e >> 3) << 4);
            qa[dstep][e] = (__bf16)qr[k];
        }
    }

    v8f   o[4];
    float mrow[8], lrow[8];
    #pragma unroll
    for (int j = 0; j < 4; ++j)
        #pragma unroll
        for (int r = 0; r < 8; ++r) o[j][r] = 0.0f;
    #pragma unroll
    for (int r = 0; r < 8; ++r) { mrow[r] = -INFINITY; lrow[r] = 0.0f; }

    const int ntiles = (blk_qb + BM - 1) / 32 + 1;
    const int srow = tid >> 2;
    const int scol = (tid & 3) << 4;

    for (int t = 0; t < ntiles; ++t) {
        const int kb = t * 32;
        {
            const float* kr = Kb + (size_t)(kb + srow) * DH + scol;
            const float* vr = Vb + (size_t)(kb + srow) * DH + scol;
            v8bf klo, khi;
            #pragma unroll
            for (int j2 = 0; j2 < 8; ++j2) {
                klo[j2] = (__bf16)kr[j2];
                khi[j2] = (__bf16)kr[8 + j2];
            }
            *(v8bf*)(sK + srow * FKSTR + scol)     = klo;
            *(v8bf*)(sK + srow * FKSTR + scol + 8) = khi;
            #pragma unroll
            for (int i = 0; i < 16; ++i)
                sVT[(scol + i) * FVSTR + srow] = (__bf16)vr[i];
        }
        __syncthreads();

        if (kb <= qb + 15) {
            v8f sc[2];
            #pragma unroll
            for (int c = 0; c < 2; ++c) {
                v8f acc;
                #pragma unroll
                for (int r = 0; r < 8; ++r) acc[r] = 0.0f;
                #pragma unroll
                for (int dstep = 0; dstep < 2; ++dstep) {
                    const __bf16* kp = sK + (16 * c + bn) * FKSTR + dstep * 32 + bk;
                    v8bf b0 = *(const v8bf*)kp;
                    v8bf b1 = *(const v8bf*)(kp + 16);
                    v16bf bf = __builtin_shufflevector(b0, b1,
                        0,1,2,3,4,5,6,7,8,9,10,11,12,13,14,15);
                    acc = __builtin_amdgcn_wmma_f32_16x16x32_bf16(
                        false, qa[dstep], false, bf, (short)0, acc, false, false);
                }
                const int key = kb + 16 * c + bn;
                #pragma unroll
                for (int r = 0; r < 8; ++r)
                    if (key > qb + cm0 + r) acc[r] = -INFINITY;
                sc[c] = acc;
            }

            float mnew[8], alpha[8], psum[8];
            #pragma unroll
            for (int r = 0; r < 8; ++r) {
                float v = redmax16(fmaxf(sc[0][r], sc[1][r]));
                mnew[r]  = fmaxf(mrow[r], v);
                alpha[r] = exp2f((mrow[r] - mnew[r]) * cf);
            }
            float* pwf = sP + wid * 16 * FPSTR;
            #pragma unroll
            for (int r = 0; r < 8; ++r) {
                float p0 = exp2f((sc[0][r] - mnew[r]) * cf);
                float p1 = exp2f((sc[1][r] - mnew[r]) * cf);
                pwf[(cm0 + r) * FPSTR + bn]      = p0;
                pwf[(cm0 + r) * FPSTR + 16 + bn] = p1;
                psum[r] = redsum16(p0 + p1);
            }
            #pragma unroll
            for (int r = 0; r < 8; ++r) {
                lrow[r] = lrow[r] * alpha[r] + psum[r];
                mrow[r] = mnew[r];
            }
            #pragma unroll
            for (int j = 0; j < 4; ++j)
                #pragma unroll
                for (int r = 0; r < 8; ++r) o[j][r] *= alpha[r];

            asm volatile("s_wait_dscnt 0" ::: "memory");

            v16bf pa;
            {
                const float* pr = pwf + am * FPSTR;
                #pragma unroll
                for (int e = 0; e < 16; ++e) {
                    const int k = ah + (e & 7) + ((e >> 3) << 4);
                    pa[e] = (__bf16)pr[k];
                }
            }
            #pragma unroll
            for (int j = 0; j < 4; ++j) {
                const __bf16* vp = sVT + (16 * j + bn) * FVSTR + bk;
                v8bf b0 = *(const v8bf*)vp;
                v8bf b1 = *(const v8bf*)(vp + 16);
                v16bf vb = __builtin_shufflevector(b0, b1,
                    0,1,2,3,4,5,6,7,8,9,10,11,12,13,14,15);
                o[j] = __builtin_amdgcn_wmma_f32_16x16x32_bf16(
                    false, pa, false, vb, (short)0, o[j], false, false);
            }
        }
        __syncthreads();
    }

    #pragma unroll
    for (int j = 0; j < 4; ++j)
        #pragma unroll
        for (int r = 0; r < 8; ++r)
            Ob[(size_t)(qb + cm0 + r) * DH + 16 * j + bn] = o[j][r] / lrow[r];
}

// ---------------- host entry ------------------------------------------------
extern "C" void kernel_launch(void* const* d_in, const int* in_sizes, int n_in,
                              void* d_out, int out_size, void* d_ws, size_t ws_size,
                              hipStream_t stream) {
    const float* q = (const float*)d_in[0];
    const float* k = (const float*)d_in[1];
    const float* v = (const float*)d_in[2];
    float*       o = (float*)d_out;
    (void)in_sizes; (void)n_in; (void)out_size;

    const size_t elems = (size_t)BH * SEQ * DH;          // 8.4M
    const size_t need  = 2 * elems * sizeof(__bf16);     // K bf16 + VT bf16

    if (ws_size >= need) {
        __bf16* kbf = (__bf16*)d_ws;
        __bf16* vt  = kbf + elems;
        conv_k <<<dim3((unsigned)(elems / (256 * 8))), dim3(256), 0, stream>>>(k, kbf);
        conv_vt<<<dim3(SEQ / 64, BH), dim3(256), 0, stream>>>(v, vt);
        fa_fwd_l2<<<dim3(SEQ / BM, BH), dim3(32 * NW), 0, stream>>>(q, kbf, vt, o);
    } else {
        fa_fwd_fallback<<<dim3(SEQ / BM, BH), dim3(32 * NW), 0, stream>>>(q, k, v, o);
    }
}
